// MoELayer_7490422964895
// MI455X (gfx1250) — matrix operand order repaired and checked by reading, not compiled, not run on hardware
//
#include <hip/hip_runtime.h>
#include <hip/hip_bf16.h>

#define N_TOKENS  8192
#define D_MODEL   1024
#define D_FF      4096
#define N_EXPERTS 8

#define TM 16
#define TN 128
#define TK 64
#define KSTEPS (TK / 32)
#define MAX_TILES (N_TOKENS / TM + N_EXPERTS)   // 520 worst case

typedef __attribute__((ext_vector_type(16))) __bf16 v16bf;
typedef __attribute__((ext_vector_type(4)))  __bf16 v4bf;
typedef __attribute__((ext_vector_type(8)))  float  v8f;

__device__ __forceinline__ __bf16 f32_to_bf16(float f) {
    union { float f; unsigned u; } in; in.f = f;
    unsigned u = in.u;
    u += 0x7FFFu + ((u >> 16) & 1u);            // round-to-nearest-even
    union { unsigned short s; __bf16 b; } out;
    out.s = (unsigned short)(u >> 16);
    return out.b;
}

// ---------------- kernel 0: zero counts + cursors ----------------
__global__ void moe_init_meta(int* ws) {
    if (threadIdx.x < 16) ws[threadIdx.x] = 0;  // counts[0..7], cursors[8..15]
}

// ---------------- kernel 1: gate logits + argmax (1 wave per token) ----------------
__global__ __launch_bounds__(256) void moe_gate(const float* __restrict__ x,
                                                const float* __restrict__ gw,
                                                const float* __restrict__ gb,
                                                int* __restrict__ top1,
                                                int* __restrict__ counts) {
    const int wv   = threadIdx.x >> 5;
    const int lane = threadIdx.x & 31;
    const int token = blockIdx.x * 8 + wv;
    if (token >= N_TOKENS) return;

    float acc[N_EXPERTS];
#pragma unroll
    for (int e = 0; e < N_EXPERTS; ++e) acc[e] = 0.0f;

    const float* xr = x + (size_t)token * D_MODEL;
    for (int d = lane; d < D_MODEL; d += 32) {
        const float xv = xr[d];
        const float* g = gw + (size_t)d * N_EXPERTS;
#pragma unroll
        for (int e = 0; e < N_EXPERTS; ++e) acc[e] += xv * g[e];
    }
#pragma unroll
    for (int off = 16; off; off >>= 1) {
#pragma unroll
        for (int e = 0; e < N_EXPERTS; ++e)
            acc[e] += __shfl_xor(acc[e], off, 32);
    }
    if (lane == 0) {
        int best = 0;
        float bv = acc[0] + gb[0];
#pragma unroll
        for (int e = 1; e < N_EXPERTS; ++e) {
            float v = acc[e] + gb[e];
            if (v > bv) { bv = v; best = e; }   // strict '>' == first-max, matches jnp.argmax
        }
        top1[token] = best;
        atomicAdd(&counts[best], 1);
    }
}

// ---------------- kernel 2: offsets + tile table (trivial, 1 thread) ----------------
__global__ void moe_build_tiles(int* ws) {
    if (threadIdx.x != 0) return;
    const int* counts   = ws;
    int* offsets        = ws + 16;              // [9]
    int* totalTiles     = ws + 25;
    int* tileExpert     = ws + 32;
    int* tileRowBase    = ws + 32 + MAX_TILES;
    int* tileRows       = ws + 32 + 2 * MAX_TILES;
    int off = 0, t = 0;
    for (int e = 0; e < N_EXPERTS; ++e) {
        offsets[e] = off;
        const int c = counts[e];
        for (int i = 0; i < c; i += TM) {
            tileExpert[t]  = e;
            tileRowBase[t] = off + i;
            tileRows[t]    = (c - i < TM) ? (c - i) : TM;
            ++t;
        }
        off += c;
    }
    offsets[N_EXPERTS] = off;
    *totalTiles = t;
}

// ---------------- kernel 3: scatter tokens into per-expert permutation ----------------
__global__ __launch_bounds__(256) void moe_scatter(const int* __restrict__ top1,
                                                   const int* __restrict__ offsets,
                                                   int* __restrict__ cursors,
                                                   int* __restrict__ perm) {
    const int n = blockIdx.x * blockDim.x + threadIdx.x;
    if (n >= N_TOKENS) return;
    const int e = top1[n];
    const int pos = atomicAdd(&cursors[e], 1);
    perm[offsets[e] + pos] = n;
}

// ---------------- kernel 4: grouped GEMM, bf16 WMMA, f32 accumulate ----------------
// LDS tiles are stored directly in WMMA *fragment layout*:
//   aFrag[s][lane][i] : lane = half*16 + row, A elem i -> K = i+16*half
//   bFrag[s][wv][lane][i] : lane = half*16 + col%16, B elem i -> K = i+16*half
// so each lane reads its whole 32B fragment with ds_load_b128s (no scalar packs).
__global__ __launch_bounds__(256) void moe_expert_gemm(
        const float* __restrict__ x,
        const float* __restrict__ ew,        // [E][D_MODEL][D_FF]
        const float* __restrict__ eb,        // [E][D_FF]
        const int*   __restrict__ perm,
        const int*   __restrict__ tileExpert,
        const int*   __restrict__ tileRowBase,
        const int*   __restrict__ tileRows,
        const int*   __restrict__ totalTiles,
        float*       __restrict__ out) {

    __shared__ alignas(32) __bf16 aFrag[KSTEPS][32][16];      // 2 KB
    __shared__ alignas(32) __bf16 bFrag[KSTEPS][8][32][16];   // 16 KB
    __shared__ int sTok[TM];

    const int t = blockIdx.x;
    if (t >= *totalTiles) return;

    const int e       = tileExpert[t];
    const int rowBase = tileRowBase[t];
    const int rows    = tileRows[t];
    const int nbase   = blockIdx.y * TN;
    const int tid     = threadIdx.x;
    const int lane    = tid & 31;
    const int wv      = tid >> 5;            // 8 waves, each owns 16 cols
    const int half    = lane >> 4;
    const int r       = lane & 15;

    if (tid < TM) sTok[tid] = (tid < rows) ? perm[rowBase + tid] : -1;
    __syncthreads();

    // ---- precomputed staging coordinates ----
    // A: thread -> (row m, 4 consecutive k at c); same half/s, consecutive elems i0..i0+3
    const int am  = tid >> 4;                 // 0..15
    const int ac  = (tid & 15) * 4;           // 0..60 (mult of 4)
    const int as  = ac >> 5;                  // kstep
    const int akk = ac & 31;
    const int ah  = akk >> 4;                 // half
    const int ai0 = akk & 15;
    const int atok0 = sTok[am];
    const int atok  = (atok0 < 0) ? 0 : atok0;   // padded rows: load row 0, never stored

    const float* wExp = ew + (size_t)e * D_MODEL * D_FF;
    v8f acc = {};

    for (int k0 = 0; k0 < D_MODEL; k0 += TK) {
        // ---- stage A: 16x64, one float4 + one v4bf LDS store per thread ----
        {
            const float4 v = *(const float4*)(x + (size_t)atok * D_MODEL + k0 + ac);
            v4bf p;
            p[0] = f32_to_bf16(v.x); p[1] = f32_to_bf16(v.y);
            p[2] = f32_to_bf16(v.z); p[3] = f32_to_bf16(v.w);
            *(v4bf*)&aFrag[as][ah * 16 + am][ai0] = p;
        }
        // ---- stage B: 64x128 in 4(k)x4(n) blocks; coalesced float4 loads,
        //      4 k-values per n land on consecutive fragment elems -> v4bf stores ----
#pragma unroll
        for (int it = 0; it < 2; ++it) {
            const int b   = tid + 256 * it;   // 0..511 blocks of 4x4
            const int kb  = (b >> 5) * 4;     // 0..60
            const int nb  = (b & 31) * 4;     // 0..124
            const int bs  = kb >> 5;
            const int bkk = kb & 31;
            const int bh  = bkk >> 4;         // half
            const int bi0 = bkk & 15;         // mult of 4
            const int bwv = nb >> 4;
            const int brr = nb & 15;          // 0,4,8,12

            const float* w0 = wExp + (size_t)(k0 + kb) * D_FF + nbase + nb;
            float rowv[4][4];
#pragma unroll
            for (int j = 0; j < 4; ++j) {
                const float4 v = *(const float4*)(w0 + (size_t)j * D_FF);
                rowv[j][0] = v.x; rowv[j][1] = v.y; rowv[j][2] = v.z; rowv[j][3] = v.w;
            }
#pragma unroll
            for (int n = 0; n < 4; ++n) {
                v4bf p;
                p[0] = f32_to_bf16(rowv[0][n]); p[1] = f32_to_bf16(rowv[1][n]);
                p[2] = f32_to_bf16(rowv[2][n]); p[3] = f32_to_bf16(rowv[3][n]);
                *(v4bf*)&bFrag[bs][bwv][bh * 16 + brr + n][bi0] = p;
            }
        }
        __syncthreads();

        // ---- WMMA: whole 32B fragments per lane via wide ds loads ----
#pragma unroll
        for (int s = 0; s < KSTEPS; ++s) {
            const v16bf a = *(const v16bf*)&aFrag[s][lane][0];
            const v16bf bm = *(const v16bf*)&bFrag[s][wv][lane][0];
            acc = __builtin_amdgcn_wmma_f32_16x16x32_bf16(
                      /*neg_a=*/false, a, /*neg_b=*/false, bm,
                      /*c_mod=*/(short)0, acc,
                      /*reuse_a=*/false, /*reuse_b=*/false);
        }
        __syncthreads();
    }

    // ---- epilogue: bias + scatter rows back to token order ----
    const int col = nbase + wv * 16 + r;
    const float bias = eb[(size_t)e * D_FF + col];
#pragma unroll
    for (int j = 0; j < 8; ++j) {
        const int m = j + 8 * half;          // C/D layout: VGPR j, lanes>=16 -> M+8
        if (m < rows)
            out[(size_t)sTok[m] * D_FF + col] = acc[j] + bias;
    }
}

extern "C" void kernel_launch(void* const* d_in, const int* in_sizes, int n_in,
                              void* d_out, int out_size, void* d_ws, size_t ws_size,
                              hipStream_t stream) {
    (void)in_sizes; (void)n_in; (void)out_size; (void)ws_size;

    const float* x  = (const float*)d_in[0];   // [8192,1024]
    const float* gw = (const float*)d_in[1];   // [1024,8]
    const float* gb = (const float*)d_in[2];   // [8]
    const float* ew = (const float*)d_in[3];   // [8,1024,4096]
    const float* eb = (const float*)d_in[4];   // [8,4096]
    float* out = (float*)d_out;                // [8192,4096]

    int* ws = (int*)d_ws;
    int* counts      = ws;
    int* cursors     = ws + 8;
    int* offsets     = ws + 16;
    int* totalTiles  = ws + 25;
    int* tileExpert  = ws + 32;
    int* tileRowBase = ws + 32 + MAX_TILES;
    int* tileRows    = ws + 32 + 2 * MAX_TILES;
    int* top1        = ws + 32 + 3 * MAX_TILES;
    int* perm        = top1 + N_TOKENS;

    moe_init_meta<<<1, 32, 0, stream>>>(ws);
    moe_gate<<<N_TOKENS / 8, 256, 0, stream>>>(x, gw, gb, top1, counts);
    moe_build_tiles<<<1, 1, 0, stream>>>(ws);
    moe_scatter<<<N_TOKENS / 256, 256, 0, stream>>>(top1, offsets, cursors, perm);

    dim3 grid(MAX_TILES, D_FF / TN);
    moe_expert_gemm<<<grid, 256, 0, stream>>>(x, ew, eb, perm,
                                              tileExpert, tileRowBase, tileRows,
                                              totalTiles, out);
}
